// EncoderImageAggr_78219944395271
// MI455X (gfx1250) — compile-verified, matrix operand order checked by reading, not run on hardware
//
#include <hip/hip_runtime.h>
#include <math.h>

typedef __attribute__((ext_vector_type(16))) _Float16 v16h;
typedef __attribute__((ext_vector_type(8)))  _Float16 v8h;
typedef __attribute__((ext_vector_type(8)))  float    v8f;
typedef __attribute__((ext_vector_type(4)))  float    f32x4;

#define EMBED 1024
#define HEADS 8
#define HDIM  128
#define IMGD  2048
#define NREG  36
#define BATCHN 512
#define FFND  4096
#define MROWS (BATCHN*NREG)   /* 18432 */
#define LAYER 3               /* only last layer's output survives in the reference */

static __device__ __forceinline__ float warp_sum(float v) {
#pragma unroll
  for (int o = 16; o > 0; o >>= 1) v += __shfl_xor(v, o, 32);
  return v;
}

static __device__ __forceinline__ float block_sum(float v, float* red8) {
  int tid = threadIdx.x;
  v = warp_sum(v);
  if ((tid & 31) == 0) red8[tid >> 5] = v;
  __syncthreads();
  float r = (tid < 8) ? red8[tid] : 0.f;
  if (tid < 32) r = warp_sum(r);
  if (tid == 0) red8[0] = r;
  __syncthreads();
  float out = red8[0];
  __syncthreads();
  return out;
}

// ------------------------------------------------------------------
// WMMA GEMM: C[M,N] = A[M,K] * W[N,K]^T + bias[N], optional ReLU.
// A, W row-major f32; converted to f16 in LDS; f32 accumulate.
// Block tile 128(M) x 128(N), K-step 32, double-buffered LDS.
// 8 waves as 4(M)x2(N); each wave owns 32x64 -> 8 v_wmma per k-step,
// one barrier per k-step, next-stage global loads issued before WMMAs.
// ------------------------------------------------------------------
#define TMS 128
#define TNS 128
#define TK  32

__global__ __launch_bounds__(256) void k_gemm(
    const float* __restrict__ A, const float* __restrict__ W,
    const float* __restrict__ bias, float* __restrict__ C,
    int M, int N, int K, int relu)
{
  __shared__ __align__(16) _Float16 As[2][TMS * TK];
  __shared__ __align__(16) _Float16 Bs[2][TNS * TK];

  const int tid  = threadIdx.x;
  const int bm   = blockIdx.x * TMS;
  const int bn   = blockIdx.y * TNS;
  const int wid  = tid >> 5, lane = tid & 31;
  const int wm   = wid & 3;        // 32-row strip
  const int wn   = wid >> 2;       // 64-col strip
  const int half = lane >> 4;
  const int sr   = lane & 15;

  v8f acc[2][4] = {};

  // cooperative stage mapping: each thread stages one 16-float chunk of A and of W
  const int rs = tid >> 1;          // 0..127  tile row
  const int cs = (tid & 1) * 16;    // col 0 or 16
  const float* aPtr = A + (size_t)(bm + rs) * K + cs;
  const float* bPtr = W + (size_t)(bn + rs) * K + cs;
  const bool aValid = (bm + rs) < M;

  const f32x4 z4 = {0.f, 0.f, 0.f, 0.f};
  f32x4 ar0, ar1, ar2, ar3, br0, br1, br2, br3;

  // ---- load stage regs for k0 ----
#define LOAD_STAGE(k0)                                                     \
  {                                                                        \
    if (aValid) {                                                          \
      const f32x4* p = (const f32x4*)(aPtr + (k0));                        \
      ar0 = p[0]; ar1 = p[1]; ar2 = p[2]; ar3 = p[3];                      \
    } else { ar0 = z4; ar1 = z4; ar2 = z4; ar3 = z4; }                     \
    const f32x4* q = (const f32x4*)(bPtr + (k0));                          \
    br0 = q[0]; br1 = q[1]; br2 = q[2]; br3 = q[3];                        \
  }

  // ---- convert staged regs and store into LDS buffer `bf` ----
#define STORE_STAGE(bf)                                                    \
  {                                                                        \
    _Float16* da = &As[bf][rs * TK + cs];                                  \
    da[0] =(_Float16)ar0.x; da[1] =(_Float16)ar0.y; da[2] =(_Float16)ar0.z; da[3] =(_Float16)ar0.w; \
    da[4] =(_Float16)ar1.x; da[5] =(_Float16)ar1.y; da[6] =(_Float16)ar1.z; da[7] =(_Float16)ar1.w; \
    da[8] =(_Float16)ar2.x; da[9] =(_Float16)ar2.y; da[10]=(_Float16)ar2.z; da[11]=(_Float16)ar2.w; \
    da[12]=(_Float16)ar3.x; da[13]=(_Float16)ar3.y; da[14]=(_Float16)ar3.z; da[15]=(_Float16)ar3.w; \
    _Float16* db = &Bs[bf][rs * TK + cs];                                  \
    db[0] =(_Float16)br0.x; db[1] =(_Float16)br0.y; db[2] =(_Float16)br0.z; db[3] =(_Float16)br0.w; \
    db[4] =(_Float16)br1.x; db[5] =(_Float16)br1.y; db[6] =(_Float16)br1.z; db[7] =(_Float16)br1.w; \
    db[8] =(_Float16)br2.x; db[9] =(_Float16)br2.y; db[10]=(_Float16)br2.z; db[11]=(_Float16)br2.w; \
    db[12]=(_Float16)br3.x; db[13]=(_Float16)br3.y; db[14]=(_Float16)br3.z; db[15]=(_Float16)br3.w; \
  }

  LOAD_STAGE(0);
  STORE_STAGE(0);
  __syncthreads();

  const int nk = K / TK;
  for (int ki = 0; ki < nk; ++ki) {
    const int buf = ki & 1;
    const bool more = (ki + 1) < nk;
    if (more) LOAD_STAGE((ki + 1) * TK);   // overlap global latency with WMMAs

    const _Float16* Ab = As[buf];
    const _Float16* Bb = Bs[buf];
    // A fragments: ISA 16-bit A layout (lane K-pairs at half*8..+7 and +16)
    v16h fa[2];
#pragma unroll
    for (int fi = 0; fi < 2; ++fi) {
      const _Float16* arow = &Ab[(wm * 32 + fi * 16 + sr) * TK + half * 8];
      v8h lo = *(const v8h*)(arow);
      v8h hi = *(const v8h*)(arow + 16);
      fa[fi] = __builtin_shufflevector(lo, hi, 0,1,2,3,4,5,6,7,8,9,10,11,12,13,14,15);
    }
#pragma unroll
    for (int t = 0; t < 4; ++t) {
      const _Float16* brow = &Bb[(wn * 64 + t * 16 + sr) * TK + half * 8];
      v8h lo = *(const v8h*)(brow);
      v8h hi = *(const v8h*)(brow + 16);
      v16h fb = __builtin_shufflevector(lo, hi, 0,1,2,3,4,5,6,7,8,9,10,11,12,13,14,15);
      acc[0][t] = __builtin_amdgcn_wmma_f32_16x16x32_f16(
          false, fa[0], false, fb, (short)0, acc[0][t], false, false);
      acc[1][t] = __builtin_amdgcn_wmma_f32_16x16x32_f16(
          false, fa[1], false, fb, (short)0, acc[1][t], false, false);
    }
    if (more) STORE_STAGE(buf ^ 1);
    __syncthreads();
  }

  // ---- epilogue: C/D layout (VGPR r, lane l) -> (M = r + 8*half, N = sr) ----
#pragma unroll
  for (int fi = 0; fi < 2; ++fi) {
#pragma unroll
    for (int t = 0; t < 4; ++t) {
      int col = bn + wn * 64 + t * 16 + sr;
      float bv = bias[col];
#pragma unroll
      for (int r = 0; r < 8; ++r) {
        int row = bm + wm * 32 + fi * 16 + half * 8 + r;
        if (row < M) {
          float v = acc[fi][t][r] + bv;
          if (relu) v = fmaxf(v, 0.f);
          C[(size_t)row * N + col] = v;
        }
      }
    }
  }
#undef LOAD_STAGE
#undef STORE_STAGE
}

// ------------------------------------------------------------------
// Sinusoid table: tab[p, 2i] = sin(p*div_i), tab[p, 2i+1] = cos(p*div_i)
// ------------------------------------------------------------------
__global__ __launch_bounds__(256) void k_sintab(float* __restrict__ tab)
{
  int p = blockIdx.x, tid = threadIdx.x;
  for (int i = tid; i < EMBED / 2; i += 256) {
    float freq = __expf((2.f * (float)i) * (-9.210340371976184f / (float)EMBED));
    float a = (float)p * freq;
    tab[(size_t)p * EMBED + 2 * i]     = __sinf(a);
    tab[(size_t)p * EMBED + 2 * i + 1] = __cosf(a);
  }
}

// ------------------------------------------------------------------
// Fused relative-position attention for one (batch, head).
// Q/K/V tiles staged global->LDS with async-to-LDS copies (ASYNCcnt).
// rel_shift is identity for Q==K.
// ------------------------------------------------------------------
__global__ __launch_bounds__(256) void k_attn(
    const float* __restrict__ Q, const float* __restrict__ Km,
    const float* __restrict__ V, const float* __restrict__ rel,
    const float* __restrict__ rwb_g, const float* __restrict__ rrb_g,
    float* __restrict__ O)
{
  __shared__ __align__(16) float qs[NREG * HDIM], ks[NREG * HDIM], vs[NREG * HDIM];
  __shared__ float sc[NREG * NREG];
  __shared__ float bw[HDIM], br[HDIM];

  int n = blockIdx.x >> 3, h = blockIdx.x & 7;
  int tid = threadIdx.x;
  size_t rowBase = (size_t)n * NREG;

  // async copy of Q/K/V (no conversion needed): per-lane 16B transfers
  for (int idx = tid * 4; idx < NREG * HDIM; idx += 256 * 4) {
    int l = idx >> 7, d = idx & 127;
    size_t g = (rowBase + l) * EMBED + (size_t)h * HDIM + d;
    const float* gq = Q + g;
    const float* gk = Km + g;
    const float* gv = V + g;
    unsigned lq = (unsigned)(uintptr_t)(&qs[idx]);
    unsigned lk = (unsigned)(uintptr_t)(&ks[idx]);
    unsigned lv = (unsigned)(uintptr_t)(&vs[idx]);
    asm volatile("global_load_async_to_lds_b128 %0, %1, off" :: "v"(lq), "v"(gq) : "memory");
    asm volatile("global_load_async_to_lds_b128 %0, %1, off" :: "v"(lk), "v"(gk) : "memory");
    asm volatile("global_load_async_to_lds_b128 %0, %1, off" :: "v"(lv), "v"(gv) : "memory");
  }
  if (tid < HDIM) { bw[tid] = rwb_g[h * HDIM + tid]; br[tid] = rrb_g[h * HDIM + tid]; }
  asm volatile("s_wait_asynccnt 0x0" ::: "memory");
  __syncthreads();

  const float inv = 0.08838834764831845f; // 1/sqrt(128)
  for (int idx = tid; idx < NREG * NREG; idx += 256) {
    int qq = idx / NREG, kk = idx - qq * NREG;
    const float* rp = rel + (size_t)(qq - kk + NREG - 1) * EMBED + (size_t)h * HDIM;
    float ac = 0.f, bd = 0.f;
#pragma unroll 4
    for (int d = 0; d < HDIM; ++d) {
      float qd = qs[qq * HDIM + d];
      ac += (qd + bw[d]) * ks[kk * HDIM + d];
      bd += (qd + br[d]) * rp[d];
    }
    sc[idx] = (ac + bd) * inv;
  }
  __syncthreads();

  if (tid < NREG) {
    float mx = -1e30f;
    for (int k2 = 0; k2 < NREG; ++k2) mx = fmaxf(mx, sc[tid * NREG + k2]);
    float sum = 0.f;
    for (int k2 = 0; k2 < NREG; ++k2) { float e = __expf(sc[tid * NREG + k2] - mx); sc[tid * NREG + k2] = e; sum += e; }
    float is = 1.f / sum;
    for (int k2 = 0; k2 < NREG; ++k2) sc[tid * NREG + k2] *= is;
  }
  __syncthreads();

  for (int idx = tid; idx < NREG * HDIM; idx += 256) {
    int l = idx >> 7, d = idx & 127;
    float s = 0.f;
    for (int k2 = 0; k2 < NREG; ++k2) s += sc[l * NREG + k2] * vs[k2 * HDIM + d];
    O[(rowBase + l) * EMBED + (size_t)h * HDIM + d] = s;
  }
}

// ------------------------------------------------------------------
// O = LayerNorm(A + B) * w + b  (per-row over 1024)
// ------------------------------------------------------------------
__global__ __launch_bounds__(256) void k_add_ln(
    const float* __restrict__ A, const float* __restrict__ B,
    const float* __restrict__ w, const float* __restrict__ b,
    float* __restrict__ O)
{
  __shared__ float red[8];
  int row = blockIdx.x, tid = threadIdx.x;
  size_t base = (size_t)row * EMBED;
  float x[4]; float s = 0.f;
#pragma unroll
  for (int i = 0; i < 4; ++i) { int c = tid + i * 256; x[i] = A[base + c] + B[base + c]; s += x[i]; }
  float mean = block_sum(s, red) * (1.f / EMBED);
  float q = 0.f;
#pragma unroll
  for (int i = 0; i < 4; ++i) { float d = x[i] - mean; q += d * d; }
  float var = block_sum(q, red) * (1.f / EMBED);
  float rs = rsqrtf(var + 1e-5f);
#pragma unroll
  for (int i = 0; i < 4; ++i) { int c = tid + i * 256; O[base + c] = (x[i] - mean) * rs * w[c] + b[c]; }
}

// attention pooling: s = softmax_l(F.ap_w + ap_b); g = sum_l F*s
__global__ __launch_bounds__(256) void k_pool(
    const float* __restrict__ F, const float* __restrict__ apw,
    const float* __restrict__ apb, float* __restrict__ g)
{
  __shared__ float s[NREG];
  int b = blockIdx.x, tid = threadIdx.x, wid = tid >> 5, lane = tid & 31;
  for (int r = wid; r < NREG; r += 8) {
    float p = 0.f;
    for (int e = lane; e < EMBED; e += 32) p += F[((size_t)b * NREG + r) * EMBED + e] * apw[e];
    p = warp_sum(p);
    if (lane == 0) s[r] = p + apb[0];
  }
  __syncthreads();
  if (tid == 0) {
    float mx = -1e30f; for (int r = 0; r < NREG; ++r) mx = fmaxf(mx, s[r]);
    float sum = 0.f;   for (int r = 0; r < NREG; ++r) { s[r] = __expf(s[r] - mx); sum += s[r]; }
    float is = 1.f / sum; for (int r = 0; r < NREG; ++r) s[r] *= is;
  }
  __syncthreads();
  for (int e = tid; e < EMBED; e += 256) {
    float a = 0.f;
    for (int r = 0; r < NREG; ++r) a += F[((size_t)b * NREG + r) * EMBED + e] * s[r];
    g[(size_t)b * EMBED + e] = a;
  }
}

// tanh(BN over region axis): params indexed by r = row % 36; in place
__global__ __launch_bounds__(256) void k_tanh_bn_loc(
    float* __restrict__ X, const float* __restrict__ w, const float* __restrict__ b,
    const float* __restrict__ m, const float* __restrict__ v)
{
  int row = blockIdx.x, tid = threadIdx.x;
  int r = row % NREG;
  float mm = m[r], sc = rsqrtf(v[r] + 1e-5f) * w[r], bb = b[r];
  size_t base = (size_t)row * EMBED;
#pragma unroll
  for (int i = 0; i < 4; ++i) { int c = tid + i * 256; X[base + c] = tanhf((X[base + c] - mm) * sc + bb); }
}

// tanh(BN over channel axis): params indexed by channel; in place
__global__ __launch_bounds__(256) void k_tanh_bn_glo(
    float* __restrict__ X, const float* __restrict__ w, const float* __restrict__ b,
    const float* __restrict__ m, const float* __restrict__ v)
{
  int bi = blockIdx.x, tid = threadIdx.x;
  size_t base = (size_t)bi * EMBED;
#pragma unroll
  for (int i = 0; i < 4; ++i) {
    int c = tid + i * 256;
    X[base + c] = tanhf((X[base + c] - m[c]) * rsqrtf(v[c] + 1e-5f) * w[c] + b[c]);
  }
}

// SGE stage 1: wts = softmax_r((l*g2).com_w + com_b); new_g = l2norm(sum_r wts*imrg)
__global__ __launch_bounds__(256) void k_sge1(
    const float* __restrict__ L, const float* __restrict__ G2,
    const float* __restrict__ cw, const float* __restrict__ cbias,
    const float* __restrict__ imrg, float* __restrict__ newg)
{
  __shared__ float s[NREG]; __shared__ float red[8];
  int b = blockIdx.x, tid = threadIdx.x, wid = tid >> 5, lane = tid & 31;
  for (int r = wid; r < NREG; r += 8) {
    float p = 0.f;
    for (int e = lane; e < EMBED; e += 32)
      p += L[((size_t)b * NREG + r) * EMBED + e] * G2[(size_t)b * EMBED + e] * cw[e];
    p = warp_sum(p);
    if (lane == 0) s[r] = p + cbias[0];
  }
  __syncthreads();
  if (tid == 0) {
    float mx = -1e30f; for (int r = 0; r < NREG; ++r) mx = fmaxf(mx, s[r]);
    float sum = 0.f;   for (int r = 0; r < NREG; ++r) { s[r] = __expf(s[r] - mx); sum += s[r]; }
    float is = 1.f / sum; for (int r = 0; r < NREG; ++r) s[r] *= is;
  }
  __syncthreads();
  float acc[8]; float ss = 0.f;
#pragma unroll
  for (int i = 0; i < 8; ++i) {
    int e = tid + i * 256;
    float a = 0.f;
    for (int r = 0; r < NREG; ++r) a += s[r] * imrg[((size_t)b * NREG + r) * IMGD + e];
    acc[i] = a; ss += a * a;
  }
  float tot = block_sum(ss, red);
  float invn = 1.f / (sqrtf(tot) + 1e-8f);
#pragma unroll
  for (int i = 0; i < 8; ++i) { int e = tid + i * 256; newg[(size_t)b * IMGD + e] = acc[i] * invn; }
}

// SGE stage 2: attn = l2norm(softmax_r(20 * <new_g, imrg_r>)); rg = imrg + attn_r * new_g
__global__ __launch_bounds__(256) void k_sge2(
    const float* __restrict__ ng, const float* __restrict__ imrg, float* __restrict__ rg)
{
  __shared__ float s[NREG];
  int b = blockIdx.x, tid = threadIdx.x, wid = tid >> 5, lane = tid & 31;
  for (int r = wid; r < NREG; r += 8) {
    float p = 0.f;
    for (int e = lane; e < IMGD; e += 32)
      p += ng[(size_t)b * IMGD + e] * imrg[((size_t)b * NREG + r) * IMGD + e];
    p = warp_sum(p);
    if (lane == 0) s[r] = p;
  }
  __syncthreads();
  if (tid == 0) {
    float mx = -1e30f; for (int r = 0; r < NREG; ++r) mx = fmaxf(mx, s[r] * 20.f);
    float sum = 0.f;
    for (int r = 0; r < NREG; ++r) { float e = __expf(s[r] * 20.f - mx); s[r] = e; sum += e; }
    float is = 1.f / sum; float ss = 0.f;
    for (int r = 0; r < NREG; ++r) { s[r] *= is; ss += s[r] * s[r]; }
    float in2 = 1.f / (sqrtf(ss) + 1e-8f);
    for (int r = 0; r < NREG; ++r) s[r] *= in2;
  }
  __syncthreads();
  size_t base = (size_t)b * NREG * IMGD;
  for (int idx = tid; idx < NREG * IMGD; idx += 256) {
    int r = idx >> 11, e = idx & (IMGD - 1);
    rg[base + idx] = imrg[base + idx] + s[r] * ng[(size_t)b * IMGD + e];
  }
}

// final per-row L2 normalization (in place on d_out)
__global__ __launch_bounds__(256) void k_l2rows(float* __restrict__ X)
{
  __shared__ float red[8];
  int row = blockIdx.x, tid = threadIdx.x;
  size_t base = (size_t)row * EMBED;
  float x[4]; float ss = 0.f;
#pragma unroll
  for (int i = 0; i < 4; ++i) { int c = tid + i * 256; x[i] = X[base + c]; ss += x[i] * x[i]; }
  float tot = block_sum(ss, red);
  float inv = 1.f / (sqrtf(tot) + 1e-8f);
#pragma unroll
  for (int i = 0; i < 4; ++i) { int c = tid + i * 256; X[base + c] = x[i] * inv; }
}

extern "C" void kernel_launch(void* const* d_in, const int* in_sizes, int n_in,
                              void* d_out, int out_size, void* d_ws, size_t ws_size,
                              hipStream_t stream)
{
  (void)in_sizes; (void)n_in; (void)out_size; (void)ws_size;
  const float* imrg  = (const float*)d_in[0];
  // d_in[1] = image_lengths (unused by the reference output)
  const float* fc_w  = (const float*)d_in[2];
  const float* fc_b  = (const float*)d_in[3];
  const float* ap_w  = (const float*)d_in[4];
  const float* ap_b  = (const float*)d_in[5];
  const float* q_w   = (const float*)d_in[6]  + (size_t)LAYER * EMBED * EMBED;
  const float* q_b   = (const float*)d_in[7]  + LAYER * EMBED;
  const float* k_w   = (const float*)d_in[8]  + (size_t)LAYER * EMBED * EMBED;
  const float* k_b   = (const float*)d_in[9]  + LAYER * EMBED;
  const float* v_w   = (const float*)d_in[10] + (size_t)LAYER * EMBED * EMBED;
  const float* v_b   = (const float*)d_in[11] + LAYER * EMBED;
  const float* o_w   = (const float*)d_in[12] + (size_t)LAYER * EMBED * EMBED;
  const float* o_b   = (const float*)d_in[13] + LAYER * EMBED;
  const float* pos_w = (const float*)d_in[14] + (size_t)LAYER * EMBED * EMBED;
  const float* pos_b = (const float*)d_in[15] + LAYER * EMBED;
  const float* rwb   = (const float*)d_in[16] + LAYER * HEADS * HDIM;
  const float* rrb   = (const float*)d_in[17] + LAYER * HEADS * HDIM;
  const float* ln1w  = (const float*)d_in[18] + LAYER * EMBED;
  const float* ln1b  = (const float*)d_in[19] + LAYER * EMBED;
  const float* ln2w  = (const float*)d_in[20] + LAYER * EMBED;
  const float* ln2b  = (const float*)d_in[21] + LAYER * EMBED;
  const float* f1w   = (const float*)d_in[22] + (size_t)LAYER * FFND * EMBED;
  const float* f1b   = (const float*)d_in[23] + LAYER * FFND;
  const float* f2w   = (const float*)d_in[24] + (size_t)LAYER * EMBED * FFND;
  const float* f2b   = (const float*)d_in[25] + LAYER * EMBED;
  const float* loc_w = (const float*)d_in[26];
  const float* loc_b = (const float*)d_in[27];
  const float* lbn_w = (const float*)d_in[28];
  const float* lbn_b = (const float*)d_in[29];
  const float* lbn_m = (const float*)d_in[30];
  const float* lbn_v = (const float*)d_in[31];
  const float* glo_w = (const float*)d_in[32];
  const float* glo_b = (const float*)d_in[33];
  const float* gbn_w = (const float*)d_in[34];
  const float* gbn_b = (const float*)d_in[35];
  const float* gbn_m = (const float*)d_in[36];
  const float* gbn_v = (const float*)d_in[37];
  const float* com_w = (const float*)d_in[38];
  const float* com_b = (const float*)d_in[39];

  float* out = (float*)d_out;
  float* ws  = (float*)d_ws;
  const size_t S = (size_t)MROWS * EMBED;

  // workspace layout (with reuse):
  float* r0 = ws;               // images -> f2 -> l_sge
  float* r1 = ws + S;           // q -> attnproj -> features
  float* r2 = ws + 2 * S;       // k -> h
  float* r3 = ws + 3 * S;       // v
  float* r4 = ws + 4 * S;       // attn_out
  float* f1buf = ws + 3 * S;    // FFN hidden (4S), overlaps dead v/attn
  float* rgbuf = ws + 2 * S;    // rg (2S), overlaps dead h/FFN
  float* tab   = ws + 7 * S;
  float* rel   = tab + (size_t)72 * EMBED;
  float* gvec  = rel + (size_t)72 * EMBED;
  float* g2    = gvec + (size_t)BATCHN * EMBED;
  float* newg  = g2 + (size_t)BATCHN * EMBED;

  dim3 blk(256);
  const int MB = MROWS / TMS;     // 144

  // 1) sinusoid table (71 x 1024)
  k_sintab<<<71, blk, 0, stream>>>(tab);
  // 2) relrow[j] = tab[j] @ pos_w^T + pos_b   (j = q-k+35)
  k_gemm<<<dim3(1, EMBED / TNS), blk, 0, stream>>>(tab, pos_w, pos_b, rel, 71, EMBED, EMBED, 0);
  // 3) images = imrg @ fc_w^T + fc_b
  k_gemm<<<dim3(MB, EMBED / TNS), blk, 0, stream>>>(imrg, fc_w, fc_b, r0, MROWS, EMBED, IMGD, 0);
  // 4) Q, K, V projections (layer 3)
  k_gemm<<<dim3(MB, EMBED / TNS), blk, 0, stream>>>(r0, q_w, q_b, r1, MROWS, EMBED, EMBED, 0);
  k_gemm<<<dim3(MB, EMBED / TNS), blk, 0, stream>>>(r0, k_w, k_b, r2, MROWS, EMBED, EMBED, 0);
  k_gemm<<<dim3(MB, EMBED / TNS), blk, 0, stream>>>(r0, v_w, v_b, r3, MROWS, EMBED, EMBED, 0);
  // 5) relative-position attention core
  k_attn<<<BATCHN * HEADS, blk, 0, stream>>>(r1, r2, r3, rel, rwb, rrb, r4);
  // 6) output projection
  k_gemm<<<dim3(MB, EMBED / TNS), blk, 0, stream>>>(r4, o_w, o_b, r1, MROWS, EMBED, EMBED, 0);
  // 7) h = LN(attnproj + images)
  k_add_ln<<<MROWS, blk, 0, stream>>>(r1, r0, ln1w, ln1b, r2);
  // 8) FFN
  k_gemm<<<dim3(MB, FFND / TNS), blk, 0, stream>>>(r2, f1w, f1b, f1buf, MROWS, FFND, EMBED, 1);
  k_gemm<<<dim3(MB, EMBED / TNS), blk, 0, stream>>>(f1buf, f2w, f2b, r0, MROWS, EMBED, FFND, 0);
  // 9) features = LN(f2 + h)
  k_add_ln<<<MROWS, blk, 0, stream>>>(r0, r2, ln2w, ln2b, r1);
  // 10) attention pooling -> g
  k_pool<<<BATCHN, blk, 0, stream>>>(r1, ap_w, ap_b, gvec);
  // 11) SGE local branch: l = tanh(BN(imrg @ loc_w^T + loc_b))
  k_gemm<<<dim3(MB, EMBED / TNS), blk, 0, stream>>>(imrg, loc_w, loc_b, r0, MROWS, EMBED, IMGD, 0);
  k_tanh_bn_loc<<<MROWS, blk, 0, stream>>>(r0, lbn_w, lbn_b, lbn_m, lbn_v);
  // 12) SGE global branch: g2 = tanh(BN(g @ glo_w^T + glo_b))
  k_gemm<<<dim3(BATCHN / TMS, EMBED / TNS), blk, 0, stream>>>(gvec, glo_w, glo_b, g2, BATCHN, EMBED, EMBED, 0);
  k_tanh_bn_glo<<<BATCHN, blk, 0, stream>>>(g2, gbn_w, gbn_b, gbn_m, gbn_v);
  // 13) SGE: weights + new_g, then region attention + rg
  k_sge1<<<BATCHN, blk, 0, stream>>>(r0, g2, com_w, com_b, imrg, newg);
  k_sge2<<<BATCHN, blk, 0, stream>>>(newg, imrg, rgbuf);
  // 14) feat = rg @ fc_w^T + fc_b  (directly into d_out), then L2-normalize rows
  k_gemm<<<dim3(MB, EMBED / TNS), blk, 0, stream>>>(rgbuf, fc_w, fc_b, out, MROWS, EMBED, IMGD, 0);
  k_l2rows<<<MROWS, blk, 0, stream>>>(out);
}
